// ExtractorMLP_41841571397745
// MI455X (gfx1250) — matrix operand order repaired and checked by reading, not compiled, not run on hardware
//
#include <hip/hip_runtime.h>
#include <hip/hip_bf16.h>
#include <stdint.h>

// ---------------------------------------------------------------------------
// ExtractorMLP on MI455X (gfx1250), bf16 WMMA pipeline.
//
// Bandwidth-bound problem (~79 GFLOP vs ~1.2 GB HBM traffic @ 23.3 TB/s).
// Intermediates stored bf16; GEMMs use v_wmma_f32_16x16x32_bf16 with fp32
// accumulation. Instance-norm stats accumulated in registers per lane
// (D-matrix layout: one column per lane) across a persistent grid-stride
// loop, flushed once via shfl_xor(16) + f32 atomics.
//
// This revision: use clang-native __bf16 converts (fptrunc/fpext) so the
// epilogue/staging conversions lower to v_cvt_*bf16* instead of the 3-op
// integer RNE emulation seen in the previous round's VALU-heavy asm.
// ---------------------------------------------------------------------------

typedef __attribute__((ext_vector_type(16))) __bf16 v16bf;
typedef __attribute__((ext_vector_type(8)))  float  v8f;

union Frag32B { uint32_t u[8]; v16bf v; };   // 32 bytes: one WMMA A/B operand

#define IN_EPS 1e-5f

__device__ __forceinline__ uint16_t f2bf(float f) {
  return __builtin_bit_cast(uint16_t, (__bf16)f);   // native cvt, RNE
}
__device__ __forceinline__ float bf2f(uint16_t h) {
  return (float)__builtin_bit_cast(__bf16, h);      // native fpext
}

// ---------------------------------------------------------------------------
// prep: W1 (128x256) and W2 (256x64) fp32 -> bf16, pre-swizzled into the
// CDNA5 WMMA B-fragment layout:
//   fragment (kt, nt): lane l holds column n = nt*16 + (l&15),
//   K values k = kt*32 + (l>>4)*8 + (e&7) + (e>>3)*16   for e = 0..15.
// Also zeroes the stats accumulators (must happen every launch).
// ---------------------------------------------------------------------------
__global__ void prep_kernel(const float* __restrict__ W1,
                            const float* __restrict__ W2,
                            uint16_t* __restrict__ W1f,
                            uint16_t* __restrict__ W2f,
                            float* __restrict__ stats1,
                            float* __restrict__ stats2) {
  const int tid = blockIdx.x * blockDim.x + threadIdx.x;
  if (tid < 32768) {                       // W1: 4 ktiles x 16 ntiles
    const int e  = tid & 15;
    const int l  = (tid >> 4) & 31;
    const int nt = (tid >> 9) & 15;
    const int kt = tid >> 13;
    const int k  = kt * 32 + ((l >> 4) * 8) + (e & 7) + ((e >> 3) * 16);
    const int n  = nt * 16 + (l & 15);
    W1f[tid] = f2bf(W1[k * 256 + n]);
  } else if (tid < 49152) {                // W2: 8 ktiles x 4 ntiles
    const int t  = tid - 32768;
    const int e  = t & 15;
    const int l  = (t >> 4) & 31;
    const int nt = (t >> 9) & 3;
    const int kt = t >> 11;
    const int k  = kt * 32 + ((l >> 4) * 8) + (e & 7) + ((e >> 3) * 16);
    const int n  = nt * 16 + (l & 15);
    W2f[t] = f2bf(W2[k * 64 + n]);
  } else if (tid < 49152 + 512) {
    stats1[tid - 49152] = 0.f;
  } else if (tid < 49152 + 512 + 128) {
    stats2[tid - 49152 - 512] = 0.f;
  }
}

// ---------------------------------------------------------------------------
// GEMM1: h1[e, 0:256] = concat(emb[col[e]], emb[row[e]]) @ W1 + b1   (bf16 out)
// Block: 256 threads = 8 waves; block tile = 32 edges x 256 cols.
// Wave (ms = w>>2, nq = w&3): 16 edges x 64 cols -> 4 accum tiles.
// Persistent grid-stride; per-lane running sum/sumsq for 4 fixed channels.
// ---------------------------------------------------------------------------
__global__ void __launch_bounds__(256)
gemm1_kernel(const float* __restrict__ emb, const int* __restrict__ eidx,
             long long E, const uint16_t* __restrict__ W1f,
             const float* __restrict__ b1,
             uint16_t* __restrict__ h1, float* __restrict__ stats1) {
  __shared__ uint16_t Abuf[2][16][128];   // 8 KB  A staging (bf16)
  __shared__ uint16_t Obuf[2][16][256];   // 16 KB output transpose staging

  const int lane = threadIdx.x & 31;
  const int w    = threadIdx.x >> 5;
  const int ms   = w >> 2;
  const int nq   = w & 3;
  const int half = lane >> 4;
  const int m    = lane & 15;

  float runS[4] = {0.f, 0.f, 0.f, 0.f};
  float runQ[4] = {0.f, 0.f, 0.f, 0.f};

  const long long nTiles = E >> 5;        // E is a multiple of 32
  for (long long tile = blockIdx.x; tile < nTiles; tile += gridDim.x) {
    const long long eBase = tile << 5;

    // ---- stage A: gather two emb rows, fp32 -> bf16, into LDS ----
    {
      const int t = threadIdx.x;
      const int s = t >> 7, r = (t >> 3) & 15, g = t & 7;  // 64B chunk per thread
      const long long e = eBase + s * 16 + r;
      const long long src = (g < 4) ? (long long)eidx[e] : (long long)eidx[E + e];
      const float4* p = (const float4*)(emb + src * 64 + (long long)(g & 3) * 16);
      uint16_t* dst = &Abuf[s][r][g * 16];
#pragma unroll
      for (int i = 0; i < 4; ++i) {
        const float4 f = p[i];
        dst[i * 4 + 0] = f2bf(f.x); dst[i * 4 + 1] = f2bf(f.y);
        dst[i * 4 + 2] = f2bf(f.z); dst[i * 4 + 3] = f2bf(f.w);
      }
    }
    __syncthreads();

    // ---- K loop: 4 ktiles x 4 ntiles of v_wmma_f32_16x16x32_bf16 ----
    v8f acc[4] = {};
#pragma unroll
    for (int kt = 0; kt < 4; ++kt) {
      Frag32B a;
      const uint16_t* ap = &Abuf[ms][m][kt * 32 + half * 8];
      *(uint4*)(&a.u[0]) = *(const uint4*)(ap);        // K lo chunk
      *(uint4*)(&a.u[4]) = *(const uint4*)(ap + 16);   // K hi chunk
#pragma unroll
      for (int j = 0; j < 4; ++j) {
        const int nt = nq * 4 + j;
        Frag32B b;
        const uint4* bp = (const uint4*)(W1f + (((kt * 16 + nt) * 32 + lane) << 4));
        *(uint4*)(&b.u[0]) = bp[0];
        *(uint4*)(&b.u[4]) = bp[1];
        acc[j] = __builtin_amdgcn_wmma_f32_16x16x32_bf16(
            false, a.v, false, b.v, (short)0, acc[j], false, false);
      }
    }

    // ---- epilogue: bias, stats, transpose through LDS ----
#pragma unroll
    for (int j = 0; j < 4; ++j) {
      const int c = nq * 64 + j * 16 + m;      // channel owned by this lane
      const float bias = b1[c];
      float s = 0.f, q = 0.f;
#pragma unroll
      for (int r = 0; r < 8; ++r) {
        const float v = acc[j][r] + bias;
        s += v; q += v * v;
        Obuf[ms][r + half * 8][c] = f2bf(v);
      }
      runS[j] += s; runQ[j] += q;
    }
    __syncthreads();

    // ---- coalesced bf16 store of the 32x256 tile ----
    {
      const int t = threadIdx.x;
      const int s = t >> 7, r = (t >> 3) & 15, g = t & 7;
      const long long e = eBase + s * 16 + r;
      uint4* dst = (uint4*)(h1 + e * 256 + g * 32);
      const uint4* src = (const uint4*)&Obuf[s][r][g * 32];
#pragma unroll
      for (int i = 0; i < 4; ++i) dst[i] = src[i];
    }
    __syncthreads();
  }

  // ---- flush per-channel stats (lane pairs L / L+16 own the same column) ----
#pragma unroll
  for (int j = 0; j < 4; ++j) {
    const float s = runS[j] + __shfl_xor(runS[j], 16, 32);
    const float q = runQ[j] + __shfl_xor(runQ[j], 16, 32);
    if (lane < 16) {
      const int c = nq * 64 + j * 16 + lane;
      atomicAdd(&stats1[c], s);
      atomicAdd(&stats1[256 + c], q);
    }
  }
}

// ---------------------------------------------------------------------------
// finalize: stats (sum, sumsq) -> (mean, rstd) with biased variance.
// ---------------------------------------------------------------------------
__global__ void finalize_kernel(const float* __restrict__ stats,
                                float* __restrict__ params, int C, float invE) {
  const int c = blockIdx.x * blockDim.x + threadIdx.x;
  if (c < C) {
    const float mean = stats[c] * invE;
    const float var  = stats[C + c] * invE - mean * mean;
    params[c]     = mean;
    params[C + c] = rsqrtf(var + IN_EPS);
  }
}

// ---------------------------------------------------------------------------
// GEMM2: h2 = relu(norm(h1)) @ W2 + b2   (bf16 out, stats over h2)
// Block: 256 threads = 8 waves; block tile = 64 edges x 64 cols.
// Wave (ms = w>>1, nh = w&1): 16 edges x 32 cols -> 2 accum tiles, K=256.
// ---------------------------------------------------------------------------
__global__ void __launch_bounds__(256)
gemm2_kernel(const uint16_t* __restrict__ h1, long long E,
             const float* __restrict__ param1,    // mean1[256] | rstd1[256]
             const uint16_t* __restrict__ W2f, const float* __restrict__ b2,
             uint16_t* __restrict__ h2, float* __restrict__ stats2) {
  __shared__ uint16_t Abuf[4][16][256];   // 32 KB
  __shared__ uint16_t Obuf[4][16][64];    // 8 KB
  __shared__ float M1[256], R1[256];

  const int lane = threadIdx.x & 31;
  const int w    = threadIdx.x >> 5;
  const int ms   = w >> 1;
  const int nh   = w & 1;
  const int half = lane >> 4;
  const int m    = lane & 15;

  M1[threadIdx.x] = param1[threadIdx.x];
  R1[threadIdx.x] = param1[256 + threadIdx.x];
  __syncthreads();

  float runS[2] = {0.f, 0.f}, runQ[2] = {0.f, 0.f};

  const long long nTiles = E >> 6;        // E is a multiple of 64
  for (long long tile = blockIdx.x; tile < nTiles; tile += gridDim.x) {
    const long long eBase = tile << 6;

    // ---- stage A = relu((h1 - mean1) * rstd1) in bf16 ----
    {
      const int t = threadIdx.x;
      const int s = t >> 6, r = (t >> 2) & 15, quar = t & 3;  // 128B per thread
      const long long e = eBase + s * 16 + r;
      const uint4* src = (const uint4*)(h1 + e * 256 + quar * 64);
      uint16_t* dst = &Abuf[s][r][quar * 64];
#pragma unroll
      for (int u = 0; u < 8; ++u) {
        const uint4 raw = src[u];
        const uint32_t vals[4] = {raw.x, raw.y, raw.z, raw.w};
#pragma unroll
        for (int p = 0; p < 4; ++p) {
          const int c0 = quar * 64 + u * 8 + p * 2;
          float lo = bf2f((uint16_t)(vals[p] & 0xffffu));
          float hi = bf2f((uint16_t)(vals[p] >> 16));
          lo = fmaxf((lo - M1[c0])     * R1[c0],     0.f);
          hi = fmaxf((hi - M1[c0 + 1]) * R1[c0 + 1], 0.f);
          dst[u * 8 + p * 2]     = f2bf(lo);
          dst[u * 8 + p * 2 + 1] = f2bf(hi);
        }
      }
    }
    __syncthreads();

    // ---- K loop: 8 ktiles x 2 ntiles ----
    v8f acc[2] = {};
#pragma unroll
    for (int kt = 0; kt < 8; ++kt) {
      Frag32B a;
      const uint16_t* ap = &Abuf[ms][m][kt * 32 + half * 8];
      *(uint4*)(&a.u[0]) = *(const uint4*)(ap);
      *(uint4*)(&a.u[4]) = *(const uint4*)(ap + 16);
#pragma unroll
      for (int j = 0; j < 2; ++j) {
        const int nt = nh * 2 + j;
        Frag32B b;
        const uint4* bp = (const uint4*)(W2f + (((kt * 4 + nt) * 32 + lane) << 4));
        *(uint4*)(&b.u[0]) = bp[0];
        *(uint4*)(&b.u[4]) = bp[1];
        acc[j] = __builtin_amdgcn_wmma_f32_16x16x32_bf16(
            false, a.v, false, b.v, (short)0, acc[j], false, false);
      }
    }

    // ---- epilogue: bias, stats, transpose through LDS ----
#pragma unroll
    for (int j = 0; j < 2; ++j) {
      const int c = nh * 32 + j * 16 + m;
      const float bias = b2[c];
      float s = 0.f, q = 0.f;
#pragma unroll
      for (int r = 0; r < 8; ++r) {
        const float v = acc[j][r] + bias;
        s += v; q += v * v;
        Obuf[ms][r + half * 8][c] = f2bf(v);
      }
      runS[j] += s; runQ[j] += q;
    }
    __syncthreads();

    // ---- coalesced bf16 store of the 64x64 tile ----
    {
      const int t = threadIdx.x;
      const int s = t >> 6, r = (t >> 2) & 15, quar = t & 3;
      const long long e = eBase + s * 16 + r;
      uint4* dst = (uint4*)(h2 + e * 64 + quar * 16);
      const uint4* src = (const uint4*)&Obuf[s][r][quar * 16];
      dst[0] = src[0];
      dst[1] = src[1];
    }
    __syncthreads();
  }

#pragma unroll
  for (int j = 0; j < 2; ++j) {
    const float s = runS[j] + __shfl_xor(runS[j], 16, 32);
    const float q = runQ[j] + __shfl_xor(runQ[j], 16, 32);
    if (lane < 16) {
      const int c = nh * 32 + j * 16 + lane;
      atomicAdd(&stats2[c], s);
      atomicAdd(&stats2[64 + c], q);
    }
  }
}

// ---------------------------------------------------------------------------
// final: out[e] = relu(norm(h2[e,:])) . W3 + b3     (N=1: VALU dot product)
// ---------------------------------------------------------------------------
__global__ void __launch_bounds__(256)
final_kernel(const uint16_t* __restrict__ h2, const float* __restrict__ param2,
             const float* __restrict__ W3, const float* __restrict__ b3,
             float* __restrict__ out, long long E) {
  __shared__ float M2[64], R2[64], Wv[64];
  if (threadIdx.x < 64) {
    M2[threadIdx.x] = param2[threadIdx.x];
    R2[threadIdx.x] = param2[64 + threadIdx.x];
    Wv[threadIdx.x] = W3[threadIdx.x];
  }
  __syncthreads();
  const float bias = b3[0];
  for (long long e = (long long)blockIdx.x * blockDim.x + threadIdx.x; e < E;
       e += (long long)gridDim.x * blockDim.x) {
    const uint4* p = (const uint4*)(h2 + e * 64);
    float acc = bias;
#pragma unroll
    for (int u = 0; u < 8; ++u) {
      const uint4 raw = p[u];
      const uint32_t vals[4] = {raw.x, raw.y, raw.z, raw.w};
#pragma unroll
      for (int pp = 0; pp < 4; ++pp) {
        const int c0 = u * 8 + pp * 2;
        const float lo = bf2f((uint16_t)(vals[pp] & 0xffffu));
        const float hi = bf2f((uint16_t)(vals[pp] >> 16));
        acc += fmaxf((lo - M2[c0])     * R2[c0],     0.f) * Wv[c0];
        acc += fmaxf((hi - M2[c0 + 1]) * R2[c0 + 1], 0.f) * Wv[c0 + 1];
      }
    }
    out[e] = acc;
  }
}

// ---------------------------------------------------------------------------
// launch
// ---------------------------------------------------------------------------
extern "C" void kernel_launch(void* const* d_in, const int* in_sizes, int n_in,
                              void* d_out, int out_size, void* d_ws, size_t ws_size,
                              hipStream_t stream) {
  const float* emb  = (const float*)d_in[0];
  const int*   eidx = (const int*)  d_in[1];   // per harness: integer -> const int*
  const float* W1   = (const float*)d_in[2];
  const float* b1   = (const float*)d_in[3];
  const float* W2   = (const float*)d_in[4];
  const float* b2   = (const float*)d_in[5];
  const float* W3   = (const float*)d_in[6];
  const float* b3   = (const float*)d_in[7];
  float* out = (float*)d_out;
  const long long E = (long long)in_sizes[1] / 2;

  // workspace layout (all offsets 256B-aligned)
  char* ws = (char*)d_ws;
  uint16_t* W1f   = (uint16_t*)(ws + 0);        // 64 KB
  uint16_t* W2f   = (uint16_t*)(ws + 65536);    // 32 KB
  float*    stats1 = (float*)(ws + 98304);      // 2 KB  (sum[256] | sumsq[256])
  float*    param1 = (float*)(ws + 100352);     // 2 KB  (mean[256] | rstd[256])
  float*    stats2 = (float*)(ws + 102400);     // 512 B
  float*    param2 = (float*)(ws + 102912);     // 512 B
  uint16_t* h1    = (uint16_t*)(ws + 103424);                       // E*512 B
  uint16_t* h2    = (uint16_t*)(ws + 103424 + (size_t)E * 512);     // E*128 B

  prep_kernel<<<(49792 + 255) / 256, 256, 0, stream>>>(W1, W2, W1f, W2f,
                                                       stats1, stats2);
  gemm1_kernel<<<2048, 256, 0, stream>>>(emb, eidx, E, W1f, b1, h1, stats1);
  finalize_kernel<<<1, 256, 0, stream>>>(stats1, param1, 256, 1.f / (float)E);
  gemm2_kernel<<<2048, 256, 0, stream>>>(h1, E, param1, W2f, b2, h2, stats2);
  finalize_kernel<<<1, 64, 0, stream>>>(stats2, param2, 64, 1.f / (float)E);
  final_kernel<<<2048, 256, 0, stream>>>(h2, param2, W3, b3, out, E);
}
